// LayerNormLSTM_11287174053966
// MI455X (gfx1250) — matrix operand order repaired and checked by reading, not compiled, not run on hardware
//
#include <hip/hip_runtime.h>
#include <hip/hip_bf16.h>

// ---------------------------------------------------------------------------
// LayerNorm-LSTM, L=2, B=128, T=512, H=I=1024 for MI455X (gfx1250, wave32).
// Per step/layer: gates via v_wmma_f32_16x16x32_bf16, f32 accumulate.
//  - bf16 weights converted once (33.5MB -> L2-resident across all 512 steps)
//  - A-strip staged per block into LDS in WMMA fragment order (converted once)
//  - register double-buffered B fragments (hide L2 latency behind WMMAs)
//  - split-K=2 partials (2048 waves, ~2/SIMD) summed in pointwise (determ.)
// ---------------------------------------------------------------------------

typedef __attribute__((ext_vector_type(16))) __bf16 bf16x16;
typedef __attribute__((ext_vector_type(8)))  float  f32x8;

#define LNUM 2
#define BSZ  128
#define TSZ  512
#define HSZ  1024
#define ISZ  1024
#define NG   4096   // 4*H gate width
#define KSPLIT 2
#define KHALF  512  // K per split
#define NKS    16   // KHALF/32 WMMA K-steps per wave

union Frag {
  bf16x16        v;
  unsigned short u[16];
  uint4          q[2];
};

__device__ __forceinline__ unsigned short f32_to_bf16_rne(float f) {
  unsigned int x = __float_as_uint(f);
  unsigned int r = 0x7FFFu + ((x >> 16) & 1u);
  return (unsigned short)((x + r) >> 16);
}

// --------------------------- weight conversion -----------------------------
__global__ __launch_bounds__(256) void cvt_weights_kernel(
    const float* __restrict__ Wih, const float* __restrict__ Whh,
    unsigned short* __restrict__ oih, unsigned short* __restrict__ ohh) {
  size_t i = (size_t)blockIdx.x * 256 + threadIdx.x;  // covers L*NG*1024 each
  oih[i] = f32_to_bf16_rne(Wih[i]);
  ohh[i] = f32_to_bf16_rne(Whh[i]);
}

__global__ __launch_bounds__(256) void zero_kernel(float* __restrict__ p) {
  size_t i = (size_t)blockIdx.x * 256 + threadIdx.x;
  p[i] = 0.0f;
}

// ------------------------------- GEMM --------------------------------------
// Gih = A_ih @ Wih^T, Ghh = A_hh @ Whh^T, each [128 x 4096], K=1024 split in 2.
// Wave decode (2048 waves, 256 blocks x 8 waves):
//   which = wid>>10 (ih/hh), ks = (wid>>9)&1, tm = (wid>>6)&7, tn = (wid&63)*64
// All 8 waves in a block share (which, ks, tm) -> A strip staged once in LDS.
//
// CDNA5 WMMA bf16 A layout (ISA 7.12.2): lanes 0-15 hold row m K{0-7,16-23},
// lanes 16-31 hold row m K{8-15,24-31}. LDS instead holds A with K contiguous (lane m: K0-15, m+16: K16-31):
// sA[kstep][lane][16 bf16] -> two ds_load_b128 per fragment, no VALU.
// B (32x16): lane n holds W row (tn+n), 16 contiguous bf16 (one 32B load).
__global__ __launch_bounds__(256) void lnlstm_gemm_kernel(
    const float* __restrict__ A_ih, long long a_ih_stride,   // row stride (elems)
    const float* __restrict__ A_hh,                          // row stride = HSZ
    const unsigned short* __restrict__ Wih_bf,               // [NG, 1024] bf16
    const unsigned short* __restrict__ Whh_bf,               // [NG, 1024] bf16
    float* __restrict__ Gih,                                 // [KSPLIT, B, NG]
    float* __restrict__ Ghh) {
  __shared__ unsigned short sA[NKS * 32 * 16];  // 16 KB

  const int wid   = (blockIdx.x << 3) | (threadIdx.x >> 5);  // 0..2047
  const int lane  = threadIdx.x & 31;
  const int which = wid >> 10;             // 0 = ih GEMM, 1 = hh GEMM
  const int ks    = (wid >> 9) & 1;        // K-split half
  const int tm    = (wid >> 6) & 7;        // M tile (16 rows)
  const int tn    = (wid & 63) << 6;       // N base (strips of 64)
  const int kbase = ks * KHALF;

  const float*     Abase   = which ? A_hh : A_ih;
  const long long  astride = which ? (long long)HSZ : a_ih_stride;
  const unsigned short* wbase = which ? Whh_bf : Wih_bf;
  float* out = (which ? Ghh : Gih) + (size_t)ks * BSZ * NG;

  // ---- Stage A strip (16 rows x KHALF) into LDS in fragment order --------
  {
    const int m   = threadIdx.x >> 4;    // 0..15 : row within strip
    const int ksl = threadIdx.x & 15;    // 0..15 : local k-step (32 K each)
    const float* src =
        Abase + (size_t)(tm * 16 + m) * (size_t)astride + kbase + ksl * 32;
    union U { unsigned short u[16]; uint4 q[2]; } lo, hi;
    for (int c = 0; c < 4; ++c) {        // chunk c covers K [c*8, c*8+8)
      float4 f0 = *(const float4*)(src + c * 8);
      float4 f1 = *(const float4*)(src + c * 8 + 4);
      unsigned short* d = (c >> 1) ? hi.u : lo.u;  // chunks 2,3 -> lane m+16
      const int s = (c & 1) * 8;
      d[s + 0] = f32_to_bf16_rne(f0.x);  d[s + 1] = f32_to_bf16_rne(f0.y);
      d[s + 2] = f32_to_bf16_rne(f0.z);  d[s + 3] = f32_to_bf16_rne(f0.w);
      d[s + 4] = f32_to_bf16_rne(f1.x);  d[s + 5] = f32_to_bf16_rne(f1.y);
      d[s + 6] = f32_to_bf16_rne(f1.z);  d[s + 7] = f32_to_bf16_rne(f1.w);
    }
    uint4* d0 = (uint4*)(sA + ksl * 512 + m * 16);
    uint4* d1 = (uint4*)(sA + ksl * 512 + (m + 16) * 16);
    d0[0] = lo.q[0]; d0[1] = lo.q[1];
    d1[0] = hi.q[0]; d1[1] = hi.q[1];
  }
  __syncthreads();

  const int mr = lane & 15;
  const int hl = lane >> 4;
  const unsigned short* w =
      wbase + (size_t)(tn + mr) * 1024 + kbase + hl * 16;
  const unsigned short* ap = sA + lane * 16;

  f32x8 acc0 = {}, acc1 = {}, acc2 = {}, acc3 = {};

#define LOADB(dst, p)                                    \
  do {                                                   \
    (dst##0).q[0] = *(const uint4*)((p));                \
    (dst##0).q[1] = *(const uint4*)((p) + 8);            \
    (dst##1).q[0] = *(const uint4*)((p) + 16 * 1024);    \
    (dst##1).q[1] = *(const uint4*)((p) + 16 * 1024 + 8);\
    (dst##2).q[0] = *(const uint4*)((p) + 32 * 1024);    \
    (dst##2).q[1] = *(const uint4*)((p) + 32 * 1024 + 8);\
    (dst##3).q[0] = *(const uint4*)((p) + 48 * 1024);    \
    (dst##3).q[1] = *(const uint4*)((p) + 48 * 1024 + 8);\
  } while (0)

#define WMMA4(a, b)                                                          \
  do {                                                                       \
    acc0 = __builtin_amdgcn_wmma_f32_16x16x32_bf16(false, (a).v, false,      \
               (b##0).v, (short)0, acc0, false, false);                      \
    acc1 = __builtin_amdgcn_wmma_f32_16x16x32_bf16(false, (a).v, false,      \
               (b##1).v, (short)0, acc1, false, false);                      \
    acc2 = __builtin_amdgcn_wmma_f32_16x16x32_bf16(false, (a).v, false,      \
               (b##2).v, (short)0, acc2, false, false);                      \
    acc3 = __builtin_amdgcn_wmma_f32_16x16x32_bf16(false, (a).v, false,      \
               (b##3).v, (short)0, acc3, false, false);                      \
  } while (0)

  Frag bc0, bc1, bc2, bc3;
  LOADB(bc, w);                                    // prime pipeline (ks=0)

  for (int kk = 0; kk < NKS - 1; ++kk) {
    Frag bn0, bn1, bn2, bn3;
    LOADB(bn, w + 32);                             // in flight during WMMAs
    __builtin_prefetch(w + 96, 0, 3);              // 2 K-steps ahead (L2 warm)
    Frag a;
    a.q[0] = *(const uint4*)(ap + kk * 512);
    a.q[1] = *(const uint4*)(ap + kk * 512 + 8);
    WMMA4(a, bc);
    bc0 = bn0; bc1 = bn1; bc2 = bn2; bc3 = bn3;
    w += 32;
  }
  {                                                // final K-step
    Frag a;
    a.q[0] = *(const uint4*)(ap + (NKS - 1) * 512);
    a.q[1] = *(const uint4*)(ap + (NKS - 1) * 512 + 8);
    WMMA4(a, bc);
  }
#undef LOADB
#undef WMMA4

  // C layout: VGPR r -> lanes 0-15 = row r, lanes 16-31 = row r+8
  const int col = tn + mr;
  for (int r = 0; r < 8; ++r) {
    const size_t row = (size_t)(tm * 16 + r + hl * 8);
    out[row * NG + col]      = acc0[r];
    out[row * NG + col + 16] = acc1[r];
    out[row * NG + col + 32] = acc2[r];
    out[row * NG + col + 48] = acc3[r];
  }
}

// ----------------------------- pointwise -----------------------------------
__device__ __forceinline__ float2 block_sum2(float a, float b) {
  // wave32 butterfly, then cross-wave combine in LDS
  for (int off = 16; off > 0; off >>= 1) {
    a += __shfl_xor(a, off, 32);
    b += __shfl_xor(b, off, 32);
  }
  __shared__ float2 part[8];
  const int wv = threadIdx.x >> 5;
  if ((threadIdx.x & 31) == 0) part[wv] = make_float2(a, b);
  __syncthreads();
  float sa = 0.f, sb = 0.f;
  for (int i = 0; i < 8; ++i) { sa += part[i].x; sb += part[i].y; }
  __syncthreads();
  return make_float2(sa, sb);
}

// One workgroup (256 threads) per batch row. Sums the two split-K partials,
// two LNs over 4096 gates, LSTM activations, bug-faithful c<-h, output LN.
__global__ __launch_bounds__(256) void lnlstm_point_kernel(
    const float* __restrict__ Gih, const float* __restrict__ Ghh,  // [2,B,NG]
    const float* __restrict__ b_ih, const float* __restrict__ b_hh,
    const float* __restrict__ g_ih, const float* __restrict__ be_ih,
    const float* __restrict__ g_hh, const float* __restrict__ be_hh,
    const float* __restrict__ g_ho, const float* __restrict__ be_ho,
    float* __restrict__ h_state_l,   // [B,H]: old h (== c, ref bug) -> new h
    float* __restrict__ ybase, long long y_stride) {
  const int b   = blockIdx.x;
  const int tid = threadIdx.x;
  const size_t part = (size_t)BSZ * NG;

  float vih[16], vhh[16];
  float s1 = 0.f, s2 = 0.f, t1 = 0.f, t2 = 0.f;
  for (int kk = 0; kk < 16; ++kk) {
    const int j = kk * 256 + tid;
    const size_t o = (size_t)b * NG + j;
    const float a = Gih[o] + Gih[o + part] + b_ih[j];
    const float c = Ghh[o] + Ghh[o + part] + b_hh[j];
    vih[kk] = a; vhh[kk] = c;
    s1 += a; s2 += a * a; t1 += c; t2 += c * c;
  }
  const float inv4h = 1.0f / (float)NG;
  float2 ri = block_sum2(s1, s2);
  float2 rh = block_sum2(t1, t2);
  const float mu_i = ri.x * inv4h;
  const float rs_i = rsqrtf(ri.y * inv4h - mu_i * mu_i + 1e-5f);
  const float mu_h = rh.x * inv4h;
  const float rs_h = rsqrtf(rh.y * inv4h - mu_h * mu_h + 1e-5f);

  float gate[16];
  for (int kk = 0; kk < 16; ++kk) {
    const int j = kk * 256 + tid;
    const float gi = (vih[kk] - mu_i) * rs_i * g_ih[j] + be_ih[j];
    const float gh = (vhh[kk] - mu_h) * rs_h * g_hh[j] + be_hh[j];
    gate[kk] = gi + gh;
  }

  // gate[kk]: kk 0-3 -> i, 4-7 -> f, 8-11 -> o, 12-15 -> g (h = kk*256+tid)
  float cnew[4], og[4];
  float cs = 0.f, cq = 0.f;
  for (int kk = 0; kk < 4; ++kk) {
    const int h = kk * 256 + tid;
    const float ig = 1.0f / (1.0f + __expf(-gate[kk]));
    const float fg = 1.0f / (1.0f + __expf(-gate[kk + 4]));
    const float oo = 1.0f / (1.0f + __expf(-gate[kk + 8]));
    const float gg = tanhf(gate[kk + 12]);
    const float c_old = h_state_l[(size_t)b * HSZ + h];  // ref bug: c stores h
    const float cn = fg * c_old + ig * gg;
    cnew[kk] = cn; og[kk] = oo;
    cs += cn; cq += cn * cn;
  }
  const float invh = 1.0f / (float)HSZ;
  float2 rc = block_sum2(cs, cq);
  const float mu_c = rc.x * invh;
  const float rs_c = rsqrtf(rc.y * invh - mu_c * mu_c + 1e-5f);

  for (int kk = 0; kk < 4; ++kk) {
    const int h = kk * 256 + tid;
    const float hn =
        og[kk] * tanhf((cnew[kk] - mu_c) * rs_c * g_ho[h] + be_ho[h]);
    h_state_l[(size_t)b * HSZ + h] = hn;
    if (ybase) ybase[(size_t)b * (size_t)y_stride + h] = hn;
  }
}

// Final hT / cT (== hT, ref bug) writeback.
__global__ __launch_bounds__(256) void writeback_kernel(
    const float* __restrict__ h, float* __restrict__ dst) {
  size_t i = (size_t)blockIdx.x * 256 + threadIdx.x;   // L*B*H elems
  const float v = h[i];
  dst[i] = v;                                  // hT
  dst[i + (size_t)LNUM * BSZ * HSZ] = v;       // cT
}

// ---------------------------------------------------------------------------
extern "C" void kernel_launch(void* const* d_in, const int* in_sizes, int n_in,
                              void* d_out, int out_size, void* d_ws, size_t ws_size,
                              hipStream_t stream) {
  const float* x     = (const float*)d_in[0];
  const float* W_ih  = (const float*)d_in[1];
  const float* W_hh  = (const float*)d_in[2];
  const float* b_ih  = (const float*)d_in[3];
  const float* b_hh  = (const float*)d_in[4];
  const float* g_ih  = (const float*)d_in[5];
  const float* be_ih = (const float*)d_in[6];
  const float* g_hh  = (const float*)d_in[7];
  const float* be_hh = (const float*)d_in[8];
  const float* g_ho  = (const float*)d_in[9];
  const float* be_ho = (const float*)d_in[10];
  float* y = (float*)d_out;

  // Workspace layout (~43 MB)
  unsigned short* Wih_bf = (unsigned short*)d_ws;                 // L*NG*1024
  unsigned short* Whh_bf = Wih_bf + (size_t)LNUM * NG * ISZ;      // L*NG*1024
  float* Gih    = (float*)(Whh_bf + (size_t)LNUM * NG * HSZ);     // [2,B,NG]
  float* Ghh    = Gih + (size_t)KSPLIT * BSZ * NG;                // [2,B,NG]
  float* hstate = Ghh + (size_t)KSPLIT * BSZ * NG;                // [L,B,H]

  cvt_weights_kernel<<<(LNUM * NG * ISZ) / 256, 256, 0, stream>>>(
      W_ih, W_hh, Wih_bf, Whh_bf);
  zero_kernel<<<(LNUM * BSZ * HSZ) / 256, 256, 0, stream>>>(hstate);

  float* h0 = hstate;
  float* h1 = hstate + (size_t)BSZ * HSZ;

  for (int t = 0; t < TSZ; ++t) {
    // Layer 0: A_ih = x[:, t, :] (row stride T*I), A_hh = h0
    lnlstm_gemm_kernel<<<256, 256, 0, stream>>>(
        x + (size_t)t * ISZ, (long long)TSZ * ISZ, h0,
        Wih_bf, Whh_bf, Gih, Ghh);
    lnlstm_point_kernel<<<BSZ, 256, 0, stream>>>(
        Gih, Ghh, b_ih, b_hh, g_ih, be_ih, g_hh, be_hh, g_ho, be_ho,
        h0, nullptr, 0);

    // Layer 1: A_ih = h0 (just updated), A_hh = h1; writes y[:, t, :]
    lnlstm_gemm_kernel<<<256, 256, 0, stream>>>(
        h0, (long long)HSZ, h1,
        Wih_bf + (size_t)NG * ISZ, Whh_bf + (size_t)NG * HSZ, Gih, Ghh);
    lnlstm_point_kernel<<<BSZ, 256, 0, stream>>>(
        Gih, Ghh, b_ih + NG, b_hh + NG, g_ih + NG, be_ih + NG,
        g_hh + NG, be_hh + NG, g_ho + HSZ, be_ho + HSZ,
        h1, y + (size_t)t * HSZ, (long long)TSZ * HSZ);
  }

  writeback_kernel<<<(LNUM * BSZ * HSZ) / 256, 256, 0, stream>>>(
      hstate, y + (size_t)BSZ * TSZ * HSZ);
}
